// HybridAttentionRecommendationNetwork_14551349199479
// MI455X (gfx1250) — matrix operand order repaired and checked by reading, not compiled, not run on hardware
//
#include <hip/hip_runtime.h>
#include <cstdint>

// ---------------------------------------------------------------------------
// Fused HybridAttentionRecommendationNetwork inference for gfx1250 (MI455X).
// Seq-len-1 attentions reduce exactly to their V projections (softmax over a
// singleton axis == 1), so the user/Q branch is dead and skipped. Every matmul
// runs on v_wmma_f32_16x16x32_f16; fragment gathers and staging use 128-bit
// vector memory ops throughout.
// ---------------------------------------------------------------------------

typedef __attribute__((ext_vector_type(16))) _Float16 v16h;
typedef __attribute__((ext_vector_type(8)))  float    v8f;
typedef __attribute__((ext_vector_type(4)))  float    v4f;
typedef __attribute__((ext_vector_type(2)))  float    v2f;
typedef __attribute__((ext_vector_type(4)))  _Float16 v4h;
typedef __attribute__((ext_vector_type(2)))  _Float16 v2h;
typedef __attribute__((ext_vector_type(4)))  unsigned int v4u;

#define DEVFN __device__ __forceinline__

union FragU { v4u q[2]; uint32_t u[8]; v16h h; };
union LnU  { v4u q; _Float16 h[8]; };

constexpr int LDA        = 264;                    // f16 act row stride (row = 528B, 16B-aligned)
constexpr int WBUF_ELEMS = 32768;                  // max weight tile (p_W2: 128x256)
constexpr int WBUF_BYTES = WBUF_ELEMS * 2;         // 65536
constexpr int SCAL_FLOATS = 256 + 256 + 64 + 64 + 64 + 16;  // sA,sB,sG,sBe,sW4,misc
constexpr int SCAL_BYTES  = SCAL_FLOATS * 4;       // 2880
constexpr int ACT_BYTES   = 16 * LDA * 2;          // 8448
constexpr int RES_BYTES   = 16 * 64 * 2;           // 2048
constexpr int LNB_BYTES   = 16 * 64 * 4;           // 4096
constexpr int WAVE_BYTES  = ACT_BYTES + RES_BYTES + LNB_BYTES;   // 14592
constexpr int NWAVES      = 8;
constexpr int SMEM_TOTAL  = WBUF_BYTES + SCAL_BYTES + NWAVES * WAVE_BYTES; // 185152

// ---- fragment load: A is 16x32 f16 from row-major LDS act [16][lda] --------
// ISA 7.12.2, 16-bit A 16x32: lanes 0-15 hold K {0..7,16..23}, lanes 16-31
// hold K {8..15,24..31}; both halves are contiguous 16B runs -> 2x b128.
DEVFN v16h load_a_frag(const _Float16* A, int lda, int kbase, int lane) {
  const int m  = lane & 15;
  const int hi = (lane >> 4) & 1;
  const v4u* row = (const v4u*)(A + m * lda + kbase);   // 16B units
  FragU f;
  f.q[0] = row[hi];        // dwords 4*hi   .. 4*hi+3   (K pairs 0..7  / 8..15)
  f.q[1] = row[2 + hi];    // dwords 8+4*hi .. 11+4*hi  (K pairs 16..23/24..31)
  return f.h;
}

// ---- fragment load: B is 32x16 = W^T tile from row-major f16 W [N][Kw] -----
// Lanes 0-15: K {0..7,16..23} of column n; lanes 16-31: K {8..15,24..31}.
DEVFN v16h load_b_frag(const _Float16* W, int Kw, int nbase, int kbase, int lane) {
  const int n  = nbase + (lane & 15);
  const int hi = (lane >> 4) & 1;
  const v4u* p = (const v4u*)(W + n * Kw + kbase);
  FragU f;
  f.q[0] = p[hi];
  f.q[1] = p[2 + hi];
  return f.h;
}

// ---- WMMA GEMM: acc[NT] (16x16 f32 tiles) = act[16xK] @ W[NTx16, K]^T ------
template <int NT, int KS>
DEVFN void gemm_acc(const _Float16* A, int lda, const _Float16* W, int Kw,
                    v8f* acc, int lane) {
#pragma unroll
  for (int t = 0; t < NT; ++t) {
#pragma unroll
    for (int e = 0; e < 8; ++e) acc[t][e] = 0.0f;
  }
#pragma unroll
  for (int ks = 0; ks < KS; ++ks) {
    const v16h a = load_a_frag(A, lda, ks * 32, lane);
#pragma unroll
    for (int t = 0; t < NT; ++t) {
      const v16h b = load_b_frag(W, Kw, t * 16, ks * 32, lane);
      acc[t] = __builtin_amdgcn_wmma_f32_16x16x32_f16(
          false, a, false, b, (short)0, acc[t], false, false);
    }
  }
}

// ---- C-fragment epilogues (col = lane&15, row = vgpr + 8*(lane>=16)) -------
DEVFN void store_bias(_Float16* act, int lda, int colBase, const v8f c,
                      const float* bias, int bBase, bool relu, int lane) {
  const int j = lane & 15, hi = (lane >> 4) & 1;
  const float b = bias[bBase + j];
#pragma unroll
  for (int r = 0; r < 8; ++r) {
    float v = c[r] + b;
    if (relu) v = fmaxf(v, 0.0f);
    act[(r + 8 * hi) * lda + colBase + j] = (_Float16)v;
  }
}

DEVFN void store_bias_dual(_Float16* act, int lda, _Float16* resid, int colBase,
                           const v8f c, const float* bias, int bBase, int lane) {
  const int j = lane & 15, hi = (lane >> 4) & 1;
  const float b = bias[bBase + j];
#pragma unroll
  for (int r = 0; r < 8; ++r) {
    const float v = c[r] + b;
    const int row = r + 8 * hi, col = colBase + j;
    act[row * lda + col]  = (_Float16)v;
    resid[row * 64 + col] = (_Float16)v;
  }
}

DEVFN void store_bias_f32(float* dst, int colBase, const v8f c,
                          const float* bias, int bBase, int lane) {
  const int j = lane & 15, hi = (lane >> 4) & 1;
  const float b = bias[bBase + j];
#pragma unroll
  for (int r = 0; r < 8; ++r) dst[(r + 8 * hi) * 64 + colBase + j] = c[r] + b;
}

DEVFN void store_resid(_Float16* act, int lda, _Float16* resid, int colBase,
                       const v8f c, const float* bias, int bBase, int lane) {
  const int j = lane & 15, hi = (lane >> 4) & 1;
  const float b = bias[bBase + j];
#pragma unroll
  for (int r = 0; r < 8; ++r) {
    const int row = r + 8 * hi, col = colBase + j;
    float v = fmaxf(c[r] + b, 0.0f);
    v += (float)resid[row * 64 + col];
    act[row * lda + col]  = (_Float16)v;
    resid[row * 64 + col] = (_Float16)v;
  }
}

DEVFN void store_ss(_Float16* act, int lda, int colBase, const v8f c,
                    const float* sc, const float* sh, int bBase, int lane) {
  const int j = lane & 15, hi = (lane >> 4) & 1;
  const float s = sc[bBase + j], t = sh[bBase + j];
#pragma unroll
  for (int r = 0; r < 8; ++r) {
    const float v = fmaxf(c[r] * s + t, 0.0f);
    act[(r + 8 * hi) * lda + colBase + j] = (_Float16)v;
  }
}

// ---- LayerNorm over 64 cols; one lane per row (lanes 0..15) ----------------
// Vectorized: 8x ds_load_b128 in, 8x ds_store_b128 out per row.
DEVFN void ln_from_f16(const _Float16* src, _Float16* act, int lda,
                       const float* g, const float* be, int lane) {
  if (lane < 16) {
    const int r = lane;
    const v4u* qs = (const v4u*)(src + r * 64);
    float xv[64], s = 0.0f, ss = 0.0f;
#pragma unroll
    for (int b = 0; b < 8; ++b) {
      LnU u; u.q = qs[b];
#pragma unroll
      for (int k = 0; k < 8; ++k) {
        const float x = (float)u.h[k];
        xv[b * 8 + k] = x; s += x; ss += x * x;
      }
    }
    const float m  = s * (1.0f / 64.0f);
    const float vr = ss * (1.0f / 64.0f) - m * m;
    const float rs = rsqrtf(vr + 1e-5f);
    v4u* qd = (v4u*)(act + r * lda);
#pragma unroll
    for (int b = 0; b < 8; ++b) {
      LnU u;
#pragma unroll
      for (int k = 0; k < 8; ++k)
        u.h[k] = (_Float16)((xv[b * 8 + k] - m) * rs * g[b * 8 + k] + be[b * 8 + k]);
      qd[b] = u.q;
    }
  }
}

DEVFN void ln_msa(const float* lnbuf, _Float16* resid, const float* g,
                  const float* be, _Float16* act, int lda, int lane) {
  if (lane < 16) {
    const int r = lane;
    const v4u* qr = (const v4u*)(resid + r * 64);
    float xv[64], s = 0.0f, ss = 0.0f;
#pragma unroll
    for (int b = 0; b < 8; ++b) {
      LnU u; u.q = qr[b];
#pragma unroll
      for (int k = 0; k < 8; ++k) {
        const float x = lnbuf[r * 64 + b * 8 + k] + (float)u.h[k];
        xv[b * 8 + k] = x; s += x; ss += x * x;
      }
    }
    const float m  = s * (1.0f / 64.0f);
    const float vr = ss * (1.0f / 64.0f) - m * m;
    const float rs = rsqrtf(vr + 1e-5f);
    v4u* qd = (v4u*)(act + r * lda);
    v4u* qw = (v4u*)(resid + r * 64);
#pragma unroll
    for (int b = 0; b < 8; ++b) {
      LnU u;
#pragma unroll
      for (int k = 0; k < 8; ++k)
        u.h[k] = (_Float16)((xv[b * 8 + k] - m) * rs * g[b * 8 + k] + be[b * 8 + k]);
      qd[b] = u.q;
      qw[b] = u.q;
    }
  }
}

// ---- block-cooperative staging (vectorized: float4 in, 4xf16 out) ----------
// n must be a multiple of 4; all weight matrices here are multiples of 1024.
DEVFN void cpW(_Float16* dst, const float* src, int n, int tid) {
  for (int i = tid * 4; i < n; i += 1024) {
    const v4f x = *(const v4f*)(src + i);
    v4h h;
    h[0] = (_Float16)x[0]; h[1] = (_Float16)x[1];
    h[2] = (_Float16)x[2]; h[3] = (_Float16)x[3];
    *(v4h*)(dst + i) = h;
  }
}
DEVFN void cpF(float* dst, const float* src, int n, int tid) {
  for (int i = tid; i < n; i += 256) dst[i] = src[i];
}
DEVFN void bnfold(float* sc, float* sh, const float* pb, const float* g,
                  const float* b, const float* m, const float* v, int n, int tid) {
  for (int i = tid; i < n; i += 256) {
    const float s = g[i] * rsqrtf(v[i] + 1e-5f);
    sc[i] = s;
    sh[i] = (pb[i] - m[i]) * s + b[i];
  }
}

struct Params {
  const int* item_idx; const float* item_features; const float* item_emb;
  const float* if_W; const float* if_b;
  const float* ca_iW; const float* ca_ib; const float* ca_oW; const float* ca_ob;
  const float* msa_inW[3]; const float* msa_inb[3];
  const float* msa_oW[3];  const float* msa_ob[3];
  const float* fusion_W; const float* fusion_b; const float* msa_g; const float* msa_be;
  const float* fi1_W1; const float* fi1_b1; const float* fi1_W2; const float* fi1_b2;
  const float* fi1_g;  const float* fi1_be;
  const float* fi2_W1; const float* fi2_b1; const float* fi2_W2; const float* fi2_b2;
  const float* fi2_g;  const float* fi2_be;
  const float* on_g;   const float* on_be;
  const float* p_W1; const float* p_b1;
  const float* bn1_g; const float* bn1_b; const float* bn1_m; const float* bn1_v;
  const float* p_W2; const float* p_b2;
  const float* bn2_g; const float* bn2_b; const float* bn2_m; const float* bn2_v;
  const float* p_W3; const float* p_b3;
  const float* bn3_g; const float* bn3_b; const float* bn3_m; const float* bn3_v;
  const float* p_W4; const float* p_b4;
  float* out;
};

__global__ __launch_bounds__(256, 1)
void harn_fused_kernel(Params p) {
  extern __shared__ char smem[];
  const int tid  = threadIdx.x;
  const int lane = tid & 31;
  const int wave = tid >> 5;
  const int rowBase = blockIdx.x * 128 + wave * 16;

  _Float16* wbuf = (_Float16*)smem;
  float* sA   = (float*)(smem + WBUF_BYTES);
  float* sB   = sA + 256;
  float* sG   = sB + 256;
  float* sBe  = sG + 64;
  float* sW4  = sBe + 64;
  float* sMisc = sW4 + 64;
  char* wbase = smem + WBUF_BYTES + SCAL_BYTES + wave * WAVE_BYTES;
  _Float16* act   = (_Float16*)wbase;
  _Float16* resid = (_Float16*)(wbase + ACT_BYTES);
  float*    lnbuf = (float*)(wbase + ACT_BYTES + RES_BYTES);

  v8f acc4[4]; v8f acc8[8]; v8f accB[16];

  // ===== Stage 1: if_lin = item_features @ if_W^T ===========================
  cpW(wbuf, p.if_W, 64 * 128, tid);
  cpF(sA, p.if_b, 64, tid);
  __syncthreads();
  __builtin_prefetch(p.ca_iW, 0, 1);
  // one float4 per lane per row: 128 cols / 32 lanes
  for (int r = 0; r < 16; ++r) {
    const v4f x = *(const v4f*)(p.item_features + (size_t)(rowBase + r) * 128 + lane * 4);
    v4h h;
    h[0] = (_Float16)x[0]; h[1] = (_Float16)x[1];
    h[2] = (_Float16)x[2]; h[3] = (_Float16)x[3];
    *(v4h*)(act + r * LDA + lane * 4) = h;
  }
  gemm_acc<4, 4>(act, LDA, wbuf, 128, acc4, lane);
  // item_c = [item_emb gather | if_lin + b]   (overwrites act after A-reads)
  for (int r = 0; r < 16; ++r) {
    const int idx = p.item_idx[rowBase + r];
    const v2f x = *(const v2f*)(p.item_emb + (size_t)idx * 64 + lane * 2);
    v2h h; h[0] = (_Float16)x[0]; h[1] = (_Float16)x[1];
    *(v2h*)(act + r * LDA + lane * 2) = h;
  }
#pragma unroll
  for (int t = 0; t < 4; ++t)
    store_bias(act, LDA, 64 + t * 16, acc4[t], sA, t * 16, false, lane);

  // ===== Stage 2: KV = item_c @ ca_iW^T  (seq-1 attention == V) =============
  __syncthreads();
  cpW(wbuf, p.ca_iW, 64 * 128, tid);
  cpF(sA, p.ca_ib, 64, tid);
  __syncthreads();
  gemm_acc<4, 4>(act, LDA, wbuf, 128, acc4, lane);
#pragma unroll
  for (int t = 0; t < 4; ++t)
    store_bias(act, LDA, t * 16, acc4[t], sA, t * 16, false, lane);

  // ===== Stage 3: ca = KV @ ca_oW^T =========================================
  __syncthreads();
  cpW(wbuf, p.ca_oW, 64 * 64, tid);
  cpF(sA, p.ca_ob, 64, tid);
  __syncthreads();
  gemm_acc<4, 2>(act, LDA, wbuf, 64, acc4, lane);
#pragma unroll
  for (int t = 0; t < 4; ++t)
    store_bias_dual(act, LDA, resid, t * 16, acc4[t], sA, t * 16, lane);

  // ===== Stage 4: MSA — only V projections matter ===========================
  __syncthreads();
  for (int i = 0; i < 3; ++i) {
    cpW(wbuf + i * 4096,         p.msa_inW[i] + 128 * 64, 4096, tid); // V rows
    cpW(wbuf + 12288 + i * 4096, p.msa_oW[i],             4096, tid);
    cpF(sA + i * 64, p.msa_inb[i] + 128, 64, tid);
    cpF(sB + i * 64, p.msa_ob[i],        64, tid);
  }
  __syncthreads();
  __builtin_prefetch(p.fusion_W, 0, 1);
  v8f accV[12];
#pragma unroll
  for (int i = 0; i < 3; ++i)
    gemm_acc<4, 2>(act, LDA, wbuf + i * 4096, 64, accV + 4 * i, lane);
#pragma unroll
  for (int i = 0; i < 3; ++i) {
#pragma unroll
    for (int t = 0; t < 4; ++t)   // v_i -> act cols 192..255 (scratch)
      store_bias(act, LDA, 192 + t * 16, accV[4 * i + t], sA, i * 64 + t * 16, false, lane);
    v8f accO[4];
    gemm_acc<4, 2>(act + 192, LDA, wbuf + 12288 + i * 4096, 64, accO, lane);
#pragma unroll
    for (int t = 0; t < 4; ++t)   // out_i -> concat slot
      store_bias(act, LDA, i * 64 + t * 16, accO[t], sB, i * 64 + t * 16, false, lane);
  }

  // ===== Stage 5: fused = concat @ fusion_W^T; msa = LN(fused + ca) =========
  __syncthreads();
  cpW(wbuf, p.fusion_W, 64 * 192, tid);
  cpF(sA, p.fusion_b, 64, tid);
  cpF(sG, p.msa_g, 64, tid);
  cpF(sBe, p.msa_be, 64, tid);
  __syncthreads();
  gemm_acc<4, 6>(act, LDA, wbuf, 192, acc4, lane);
#pragma unroll
  for (int t = 0; t < 4; ++t)
    store_bias_f32(lnbuf, t * 16, acc4[t], sA, t * 16, lane);
  ln_msa(lnbuf, resid, sG, sBe, act, LDA, lane);

  // ===== FI1: LN -> 64->256 relu -> 256->64 relu -> +x ======================
  __syncthreads();
  cpF(sG, p.fi1_g, 64, tid);
  cpF(sBe, p.fi1_be, 64, tid);
  cpW(wbuf, p.fi1_W1, 256 * 64, tid);
  cpF(sA, p.fi1_b1, 256, tid);
  __syncthreads();
  ln_from_f16(resid, act, LDA, sG, sBe, lane);
  gemm_acc<16, 2>(act, LDA, wbuf, 64, accB, lane);
#pragma unroll
  for (int t = 0; t < 16; ++t)
    store_bias(act, LDA, t * 16, accB[t], sA, t * 16, true, lane);
  __syncthreads();
  cpW(wbuf, p.fi1_W2, 64 * 256, tid);
  cpF(sA, p.fi1_b2, 64, tid);
  __syncthreads();
  gemm_acc<4, 8>(act, LDA, wbuf, 256, acc4, lane);
#pragma unroll
  for (int t = 0; t < 4; ++t)
    store_resid(act, LDA, resid, t * 16, acc4[t], sA, t * 16, lane);

  // ===== FI2: LN -> 64->128 relu -> 128->64 relu -> +x ======================
  __syncthreads();
  cpF(sG, p.fi2_g, 64, tid);
  cpF(sBe, p.fi2_be, 64, tid);
  cpW(wbuf, p.fi2_W1, 128 * 64, tid);
  cpF(sA, p.fi2_b1, 128, tid);
  __syncthreads();
  ln_from_f16(resid, act, LDA, sG, sBe, lane);
  gemm_acc<8, 2>(act, LDA, wbuf, 64, acc8, lane);
#pragma unroll
  for (int t = 0; t < 8; ++t)
    store_bias(act, LDA, t * 16, acc8[t], sA, t * 16, true, lane);
  __syncthreads();
  cpW(wbuf, p.fi2_W2, 64 * 128, tid);
  cpF(sA, p.fi2_b2, 64, tid);
  __syncthreads();
  gemm_acc<4, 4>(act, LDA, wbuf, 128, acc4, lane);
#pragma unroll
  for (int t = 0; t < 4; ++t)
    store_resid(act, LDA, resid, t * 16, acc4[t], sA, t * 16, lane);

  // ===== Final LN + prediction layer 1 (BN folded) ==========================
  __syncthreads();
  cpF(sG, p.on_g, 64, tid);
  cpF(sBe, p.on_be, 64, tid);
  cpW(wbuf, p.p_W1, 256 * 64, tid);
  bnfold(sA, sB, p.p_b1, p.bn1_g, p.bn1_b, p.bn1_m, p.bn1_v, 256, tid);
  __syncthreads();
  __builtin_prefetch(p.p_W2, 0, 1);
  ln_from_f16(resid, act, LDA, sG, sBe, lane);
  gemm_acc<16, 2>(act, LDA, wbuf, 64, accB, lane);
#pragma unroll
  for (int t = 0; t < 16; ++t)
    store_ss(act, LDA, t * 16, accB[t], sA, sB, t * 16, lane);

  // ===== Prediction layer 2: 256 -> 128 =====================================
  __syncthreads();
  cpW(wbuf, p.p_W2, 128 * 256, tid);
  bnfold(sA, sB, p.p_b2, p.bn2_g, p.bn2_b, p.bn2_m, p.bn2_v, 128, tid);
  __syncthreads();
  gemm_acc<8, 8>(act, LDA, wbuf, 256, acc8, lane);
#pragma unroll
  for (int t = 0; t < 8; ++t)
    store_ss(act, LDA, t * 16, acc8[t], sA, sB, t * 16, lane);

  // ===== Prediction layer 3: 128 -> 64 ======================================
  __syncthreads();
  cpW(wbuf, p.p_W3, 64 * 128, tid);
  bnfold(sA, sB, p.p_b3, p.bn3_g, p.bn3_b, p.bn3_m, p.bn3_v, 64, tid);
  __syncthreads();
  gemm_acc<4, 4>(act, LDA, wbuf, 128, acc4, lane);
#pragma unroll
  for (int t = 0; t < 4; ++t)
    store_ss(act, LDA, t * 16, acc4[t], sA, sB, t * 16, lane);

  // ===== Final dot: 64 -> 1 =================================================
  __syncthreads();
  cpF(sW4, p.p_W4, 64, tid);
  if (tid == 0) sMisc[0] = p.p_b4[0];
  __syncthreads();
  if (lane < 16) {
    const int r = lane;
    const v4u* qs = (const v4u*)(act + r * LDA);
    float s = sMisc[0];
#pragma unroll
    for (int b = 0; b < 8; ++b) {
      LnU u; u.q = qs[b];
#pragma unroll
      for (int k = 0; k < 8; ++k) s += (float)u.h[k] * sW4[b * 8 + k];
    }
    p.out[rowBase + r] = s;
  }
}

extern "C" void kernel_launch(void* const* d_in, const int* in_sizes, int n_in,
                              void* d_out, int out_size, void* d_ws, size_t ws_size,
                              hipStream_t stream) {
  (void)in_sizes; (void)n_in; (void)out_size; (void)d_ws; (void)ws_size;
  Params p;
  p.item_idx      = (const int*)d_in[4];
  p.item_features = (const float*)d_in[5];
  // params flattened in setup_inputs() insertion order, starting at index 6
  p.item_emb = (const float*)d_in[7];
  p.if_W = (const float*)d_in[12]; p.if_b = (const float*)d_in[13];
  p.ca_iW = (const float*)d_in[16]; p.ca_ib = (const float*)d_in[17];
  p.ca_oW = (const float*)d_in[18]; p.ca_ob = (const float*)d_in[19];
  for (int i = 0; i < 3; ++i) {
    p.msa_inW[i] = (const float*)d_in[20 + i];
    p.msa_inb[i] = (const float*)d_in[23 + i];
    p.msa_oW[i]  = (const float*)d_in[26 + i];
    p.msa_ob[i]  = (const float*)d_in[29 + i];
  }
  p.fusion_W = (const float*)d_in[32]; p.fusion_b = (const float*)d_in[33];
  p.msa_g = (const float*)d_in[34];    p.msa_be  = (const float*)d_in[35];
  p.fi1_W1 = (const float*)d_in[36]; p.fi1_b1 = (const float*)d_in[37];
  p.fi1_W2 = (const float*)d_in[38]; p.fi1_b2 = (const float*)d_in[39];
  p.fi1_g  = (const float*)d_in[40]; p.fi1_be = (const float*)d_in[41];
  p.fi2_W1 = (const float*)d_in[42]; p.fi2_b1 = (const float*)d_in[43];
  p.fi2_W2 = (const float*)d_in[44]; p.fi2_b2 = (const float*)d_in[45];
  p.fi2_g  = (const float*)d_in[46]; p.fi2_be = (const float*)d_in[47];
  p.on_g   = (const float*)d_in[48]; p.on_be  = (const float*)d_in[49];
  p.p_W1 = (const float*)d_in[50]; p.p_b1 = (const float*)d_in[51];
  p.bn1_g = (const float*)d_in[52]; p.bn1_b = (const float*)d_in[53];
  p.bn1_m = (const float*)d_in[54]; p.bn1_v = (const float*)d_in[55];
  p.p_W2 = (const float*)d_in[56]; p.p_b2 = (const float*)d_in[57];
  p.bn2_g = (const float*)d_in[58]; p.bn2_b = (const float*)d_in[59];
  p.bn2_m = (const float*)d_in[60]; p.bn2_v = (const float*)d_in[61];
  p.p_W3 = (const float*)d_in[62]; p.p_b3 = (const float*)d_in[63];
  p.bn3_g = (const float*)d_in[64]; p.bn3_b = (const float*)d_in[65];
  p.bn3_m = (const float*)d_in[66]; p.bn3_v = (const float*)d_in[67];
  p.p_W4 = (const float*)d_in[68]; p.p_b4 = (const float*)d_in[69];
  p.out = (float*)d_out;

  dim3 grid(256), block(256);   // 256 blocks * 8 waves * 16 rows = 32768
  harn_fused_kernel<<<grid, block, SMEM_TOTAL, stream>>>(p);
}
